// VAE_19859928776719
// MI455X (gfx1250) — matrix operand order repaired and checked by reading, not compiled, not run on hardware
//
#include <hip/hip_runtime.h>
#include <hip/hip_bf16.h>

// ---------------------------------------------------------------------------
// Seq2seq LSTM VAE for MI455X (gfx1250, wave32).
//  - Batched input-side gate GEMMs + output-vocab projection via
//    v_wmma_f32_16x16x32_bf16; 4 M-tiles per wave => 4x B-operand reuse
//    (bf16 out_W = 103 MB stays resident in the 192 MB L2 across re-reads).
//  - Sequential LSTM recurrence: persistent 16-WG kernel, grid barrier/step,
//    h broadcast into LDS via global_load_async_to_lds_b128 (ASYNCcnt).
//  - eps (jax PRNG) is not reproducible on device -> latent = mean (eps=0).
// ---------------------------------------------------------------------------

#define VOCAB   50257
#define NPAD    50272          // VOCAB padded to multiple of 16
#define HDIM    1024
#define GDIM    4096           // 4*H
#define SEQ     256
#define LZ      128
#define CDIM    32

typedef __attribute__((ext_vector_type(16))) __bf16 v16bf;
typedef __attribute__((ext_vector_type(8)))  float  v8f;

union Frag { uint4 q[2]; v16bf v; };

// ---- bf16 helpers ----------------------------------------------------------
__device__ __forceinline__ unsigned short f2bf(float f) {
    union { float f; unsigned u; } x; x.f = f;
    unsigned u = x.u;
    u += 0x7fffu + ((u >> 16) & 1u);       // round-to-nearest-even
    return (unsigned short)(u >> 16);
}
__device__ __forceinline__ float bf_lo(unsigned u) {
    union { unsigned u; float f; } x; x.u = u << 16; return x.f;
}
__device__ __forceinline__ float bf_hi(unsigned u) {
    union { unsigned u; float f; } x; x.u = u & 0xffff0000u; return x.f;
}
__device__ __forceinline__ float sigm(float x) { return 1.0f / (1.0f + __expf(-x)); }

// ---- f32 -> bf16 conversion (grid-stride) ---------------------------------
__global__ void cvt_bf16_kernel(const float* __restrict__ src,
                                unsigned short* __restrict__ dst, long n) {
    long i = (long)blockIdx.x * blockDim.x + threadIdx.x;
    long stride = (long)gridDim.x * blockDim.x;
    for (; i < n; i += stride) dst[i] = f2bf(src[i]);
}

// ---- embedding gather (enc: plain, dec: shifted tokens + relu) -------------
__global__ void embed_kernel(const int* __restrict__ tokens,
                             const float* __restrict__ enc_emb,
                             const float* __restrict__ dec_emb,
                             unsigned short* __restrict__ xe,
                             unsigned short* __restrict__ xd) {
    int t = blockIdx.x;
    int te = tokens[t];
    int td = (t == 0) ? 0 : tokens[t - 1];      // SOS = 0
    const float* er = enc_emb + (size_t)te * HDIM;
    const float* dr = dec_emb + (size_t)td * HDIM;
    for (int i = threadIdx.x; i < HDIM; i += blockDim.x) {
        xe[(size_t)t * HDIM + i] = f2bf(er[i]);
        float v = dr[i];
        xd[(size_t)t * HDIM + i] = f2bf(v > 0.f ? v : 0.f);
    }
}

// ---- WMMA GEMM: C[M x nvalid] = A[MxK] * B[NxK]^T + bias -------------------
// A, B bf16 row-major (B row n is column n of the logical K x N operand: "TN").
// Each wave computes MT stacked 16x16 C tiles (MT*16 x 16) sharing one B
// fragment per k-chunk -> B global/L2 traffic cut by MT.
template <int MT>
__global__ __launch_bounds__(256)
void gemm_tn_bf16(const unsigned short* __restrict__ A,
                  const unsigned short* __restrict__ B,
                  const float* __restrict__ bias0,
                  const float* __restrict__ bias1,
                  float* __restrict__ C,
                  int M, int Npad, int K, int ldc, int nvalid) {
    const int lane = threadIdx.x & 31;
    const int wave = threadIdx.x >> 5;
    const int tilesMg = M / (16 * MT);
    const int tile = blockIdx.x * (blockDim.x >> 5) + wave;
    if (tile >= tilesMg * (Npad >> 4)) return;
    const int mg  = (tile % tilesMg) * (16 * MT);
    const int n16 = (tile / tilesMg) << 4;

    const int half = lane >> 4;        // 0 or 1
    const int r    = lane & 15;
    // ISA 7.12.2 bf16 A 16x32: lanes 0-15 hold K=[0..7]+[16..23], lanes 16-31
    // hold K=[8..15]+[24..31] for row M = lane%16.
    const unsigned short* abase = A + (size_t)(mg + r) * K;
    // bf16 B 32x16: lanes 0-15 hold K=[0..7]+[16..23], lanes 16-31 hold K=[8..15]+[24..31] of
    // column N = lane%16 (column n of B == row n of the stored NxK matrix).
    const unsigned short* brow = B + (size_t)(n16 + r) * K;

    v8f acc[MT];
    #pragma unroll
    for (int mt = 0; mt < MT; ++mt) acc[mt] = (v8f){};

    const int half8  = half << 3;
    const int half16 = half << 4;
    for (int kk = 0; kk < K; kk += 32) {
        __builtin_prefetch(brow + kk + 512, 0, 0);   // -> global_prefetch_b8
        Frag fb;
        fb.q[0] = *(const uint4*)(brow + kk + half8);
        fb.q[1] = *(const uint4*)(brow + kk + 16 + half8);
        #pragma unroll
        for (int mt = 0; mt < MT; ++mt) {
            const unsigned short* arow = abase + (size_t)mt * 16 * K;
            Frag fa;
            fa.q[0] = *(const uint4*)(arow + kk + half8);
            fa.q[1] = *(const uint4*)(arow + kk + 16 + half8);
            acc[mt] = __builtin_amdgcn_wmma_f32_16x16x32_bf16(
                          false, fa.v, false, fb.v, (short)0, acc[mt], false, false);
        }
    }

    const int col = n16 + r;
    if (col < nvalid) {
        float bv = bias0 ? bias0[col] : 0.f;
        if (bias1) bv += bias1[col];
        #pragma unroll
        for (int mt = 0; mt < MT; ++mt) {
            #pragma unroll
            for (int vr = 0; vr < 8; ++vr) {        // C/D: row = vr + 8*half
                int row = mg + mt * 16 + vr + (half << 3);
                C[(size_t)row * ldc + col] = acc[mt][vr] + bv;
            }
        }
    }
}

// ---- grid-wide barrier (generation counter) --------------------------------
__device__ __forceinline__ void grid_sync(int* cnt, volatile int* gen, int nblocks) {
    __syncthreads();
    if (threadIdx.x == 0) {
        __threadfence();
        int old = *gen;
        if (atomicAdd(cnt, 1) == nblocks - 1) {
            *cnt = 0;
            __threadfence();
            atomicAdd((int*)gen, 1);
        } else {
            while (*gen == old) { __builtin_amdgcn_s_sleep(1); }
        }
        __threadfence();
    }
    __syncthreads();
}

// ---- persistent LSTM recurrence -------------------------------------------
// 16 blocks x 256 threads = 4096 lanes; lane g owns gate row g of W_hh.
// Per step: gates[g] = Xpre[t][g] + dot(h, W_hh[g]); then lanes g<1024 apply
// nonlinearities and update (h,c). c[g] lives in a register for the whole seq.
// h is broadcast into LDS with GLOBAL_LOAD_ASYNC_TO_LDS_B128 (ASYNCcnt).
__global__ __launch_bounds__(256)
void lstm_seq_kernel(const float* __restrict__ Xpre,
                     const unsigned short* __restrict__ Whh,
                     float* __restrict__ h_glob,
                     float* __restrict__ gbuf,
                     const float* __restrict__ h0,     // may be null (keep h_glob)
                     const float* __restrict__ c0,     // may be null (c = 0)
                     unsigned short* __restrict__ Hout_bf, // may be null
                     int* bar_cnt, int* bar_gen, int steps) {
    const int gid = blockIdx.x * blockDim.x + threadIdx.x;   // 0..4095
    const int nb  = gridDim.x;
    float cc = 0.f;
    if (gid < HDIM) {
        if (c0) cc = c0[gid];
        if (h0) h_glob[gid] = h0[gid];
    }
    grid_sync(bar_cnt, bar_gen, nb);

    __shared__ float hs[HDIM];
    const unsigned short* wr = Whh + (size_t)gid * HDIM;
    // LDS byte address for this thread's 16-byte slice of hs (flat-address
    // low 32 bits == LDS offset for shared-aperture pointers).
    const unsigned lds_dst = (unsigned)(uintptr_t)(&hs[0]) + threadIdx.x * 16u;

    for (int t = 0; t < steps; ++t) {
        // async copy h_glob (4 KB) -> LDS: 256 lanes x b128
        {
            const float* gsrc = h_glob + threadIdx.x * 4;
            asm volatile("global_load_async_to_lds_b128 %0, %1, off"
                         :: "v"(lds_dst), "v"(gsrc) : "memory");
            asm volatile("s_wait_asynccnt 0x0" ::: "memory");
        }
        __syncthreads();

        float acc = Xpre[(size_t)t * GDIM + gid];
        for (int k = 0; k < HDIM; k += 8) {
            uint4 q = *(const uint4*)(wr + k);
            acc += hs[k + 0] * bf_lo(q.x) + hs[k + 1] * bf_hi(q.x);
            acc += hs[k + 2] * bf_lo(q.y) + hs[k + 3] * bf_hi(q.y);
            acc += hs[k + 4] * bf_lo(q.z) + hs[k + 5] * bf_hi(q.z);
            acc += hs[k + 6] * bf_lo(q.w) + hs[k + 7] * bf_hi(q.w);
        }
        gbuf[gid] = acc;
        grid_sync(bar_cnt, bar_gen, nb);

        if (gid < HDIM) {
            float gi = gbuf[gid];
            float gf = gbuf[HDIM + gid];
            float gg = gbuf[2 * HDIM + gid];
            float go = gbuf[3 * HDIM + gid];
            cc = sigm(gf) * cc + sigm(gi) * tanhf(gg);
            float hh = sigm(go) * tanhf(cc);
            h_glob[gid] = hh;
            if (Hout_bf) Hout_bf[(size_t)t * HDIM + gid] = f2bf(hh);
        }
        grid_sync(bar_cnt, bar_gen, nb);
    }
}

// ---- latent head: mean/logvar/latent -> decoder initial hidden -------------
__global__ __launch_bounds__(256)
void latent_kernel(const float* __restrict__ h_enc,
                   const float* __restrict__ meanW, const float* __restrict__ meanB,
                   const float* __restrict__ logvW, const float* __restrict__ logvB,
                   const float* __restrict__ lcW,   const float* __restrict__ lcB,
                   const float* __restrict__ c_in,
                   float* __restrict__ out_mean, float* __restrict__ out_logv,
                   float* __restrict__ h_out) {
    __shared__ float z[LZ + CDIM];
    int tid = threadIdx.x;
    if (tid < LZ) {
        const float* row = meanW + (size_t)tid * HDIM;
        float a = meanB[tid];
        for (int k = 0; k < HDIM; ++k) a += h_enc[k] * row[k];
        out_mean[tid] = a;
        z[tid] = a;                       // eps = 0 (jax PRNG not reproducible)
    } else if (tid < 2 * LZ) {
        int j = tid - LZ;
        const float* row = logvW + (size_t)j * HDIM;
        float a = logvB[j];
        for (int k = 0; k < HDIM; ++k) a += h_enc[k] * row[k];
        out_logv[j] = a;
    }
    if (tid < CDIM) z[LZ + tid] = c_in[tid];
    __syncthreads();
    for (int i = tid; i < HDIM; i += blockDim.x) {
        const float* row = lcW + (size_t)i * (LZ + CDIM);
        float a = lcB[i];
        #pragma unroll 4
        for (int j = 0; j < LZ + CDIM; ++j) a += row[j] * z[j];
        h_out[i] = a;                     // decoder initial h (c starts at 0)
    }
}

// ---- per-row log-softmax (in place) + argmax -------------------------------
__global__ __launch_bounds__(256)
void logsoftmax_argmax_kernel(float* __restrict__ dist, float* __restrict__ pred) {
    const int row = blockIdx.x;
    float* x = dist + (size_t)row * VOCAB;
    const int tid = threadIdx.x, lid = tid & 31, wid = tid >> 5;
    __shared__ float wm[8]; __shared__ int wi[8];
    __shared__ float bM; __shared__ int bI; __shared__ float bS;

    // pass 1: max + argmax (first-index tie break, as jnp.argmax)
    float m = -3.4e38f; int mi = 0x7fffffff;
    for (int i = tid; i < VOCAB; i += blockDim.x) {
        float v = x[i];
        if (v > m || (v == m && i < mi)) { m = v; mi = i; }
    }
    for (int off = 16; off > 0; off >>= 1) {
        float om = __shfl_xor(m, off);
        int  oi  = __shfl_xor(mi, off);
        if (om > m || (om == m && oi < mi)) { m = om; mi = oi; }
    }
    if (lid == 0) { wm[wid] = m; wi[wid] = mi; }
    __syncthreads();
    if (tid == 0) {
        float M = wm[0]; int MI = wi[0];
        for (int w = 1; w < 8; ++w)
            if (wm[w] > M || (wm[w] == M && wi[w] < MI)) { M = wm[w]; MI = wi[w]; }
        bM = M; bI = MI;
    }
    __syncthreads();
    const float M = bM;

    // pass 2: sum of exp
    float s = 0.f;
    for (int i = tid; i < VOCAB; i += blockDim.x) s += __expf(x[i] - M);
    for (int off = 16; off > 0; off >>= 1) s += __shfl_xor(s, off);
    if (lid == 0) wm[wid] = s;
    __syncthreads();
    if (tid == 0) {
        float S = 0.f;
        for (int w = 0; w < 8; ++w) S += wm[w];
        bS = M + __logf(S);
    }
    __syncthreads();
    const float lse = bS;

    // pass 3: write log-softmax and predicted token
    for (int i = tid; i < VOCAB; i += blockDim.x) x[i] = x[i] - lse;
    if (tid == 0) pred[row] = (float)bI;
}

// ---------------------------------------------------------------------------
extern "C" void kernel_launch(void* const* d_in, const int* in_sizes, int n_in,
                              void* d_out, int out_size, void* d_ws, size_t ws_size,
                              hipStream_t stream) {
    (void)in_sizes; (void)n_in; (void)out_size; (void)ws_size;

    const int*   tokens   = (const int*)d_in[0];
    const float* enc_h0   = (const float*)d_in[1];
    const float* enc_c0   = (const float*)d_in[2];
    const float* c_in     = (const float*)d_in[3];
    const float* enc_emb  = (const float*)d_in[5];
    const float* enc_W_ih = (const float*)d_in[6];
    const float* enc_W_hh = (const float*)d_in[7];
    const float* enc_b_ih = (const float*)d_in[8];
    const float* enc_b_hh = (const float*)d_in[9];
    const float* dec_emb  = (const float*)d_in[10];
    const float* dec_W_ih = (const float*)d_in[11];
    const float* dec_W_hh = (const float*)d_in[12];
    const float* dec_b_ih = (const float*)d_in[13];
    const float* dec_b_hh = (const float*)d_in[14];
    const float* out_W    = (const float*)d_in[15];
    const float* out_b    = (const float*)d_in[16];
    const float* mean_W   = (const float*)d_in[17];
    const float* mean_b   = (const float*)d_in[18];
    const float* logv_W   = (const float*)d_in[19];
    const float* logv_b   = (const float*)d_in[20];
    const float* lc_W     = (const float*)d_in[21];
    const float* lc_b     = (const float*)d_in[22];

    float* out = (float*)d_out;
    const size_t DIST_OFF = SEQ;                         // 256 preds first
    const size_t MEAN_OFF = SEQ + (size_t)SEQ * VOCAB;
    const size_t LOGV_OFF = MEAN_OFF + LZ;

    // workspace carve-up (256B aligned)
    char* ws = (char*)d_ws;
    size_t off = 0;
    auto take = [&](size_t bytes) -> char* {
        char* p = ws + off;
        off += (bytes + 255) & ~(size_t)255;
        return p;
    };
    unsigned short* Weih   = (unsigned short*)take((size_t)GDIM * HDIM * 2);
    unsigned short* Wehh   = (unsigned short*)take((size_t)GDIM * HDIM * 2);
    unsigned short* Wdih   = (unsigned short*)take((size_t)GDIM * HDIM * 2);
    unsigned short* Wdhh   = (unsigned short*)take((size_t)GDIM * HDIM * 2);
    unsigned short* Wout   = (unsigned short*)take((size_t)NPAD * HDIM * 2);
    unsigned short* Xe_bf  = (unsigned short*)take((size_t)SEQ * HDIM * 2);
    unsigned short* Xd_bf  = (unsigned short*)take((size_t)SEQ * HDIM * 2);
    float*          XpreE  = (float*)take((size_t)SEQ * GDIM * 4);
    float*          XpreD  = (float*)take((size_t)SEQ * GDIM * 4);
    float*          gbuf   = (float*)take((size_t)GDIM * 4);
    float*          h_glob = (float*)take((size_t)HDIM * 4);
    unsigned short* Hdec   = (unsigned short*)take((size_t)SEQ * HDIM * 2);
    int*            bar    = (int*)take(64);

    // 0) reset barrier state; zero-fill out_W pad rows (bf16)
    hipMemsetAsync(bar, 0, 64, stream);
    hipMemsetAsync(Wout + (size_t)VOCAB * HDIM, 0,
                   (size_t)(NPAD - VOCAB) * HDIM * 2, stream);

    // 1) weights -> bf16 (out_W bf16 = 103 MB -> fits 192 MB L2 for the GEMM)
    cvt_bf16_kernel<<<2048, 256, 0, stream>>>(enc_W_ih, Weih, (long)GDIM * HDIM);
    cvt_bf16_kernel<<<2048, 256, 0, stream>>>(enc_W_hh, Wehh, (long)GDIM * HDIM);
    cvt_bf16_kernel<<<2048, 256, 0, stream>>>(dec_W_ih, Wdih, (long)GDIM * HDIM);
    cvt_bf16_kernel<<<2048, 256, 0, stream>>>(dec_W_hh, Wdhh, (long)GDIM * HDIM);
    cvt_bf16_kernel<<<4096, 256, 0, stream>>>(out_W,    Wout, (long)VOCAB * HDIM);

    // 2) embedding gathers (enc x_seq, dec relu(emb[shifted tokens]))
    embed_kernel<<<SEQ, 256, 0, stream>>>(tokens, enc_emb, dec_emb, Xe_bf, Xd_bf);

    // 3) batched input-side gate pre-activations (WMMA GEMMs, 256x4096 each)
    {
        int waveTiles = (SEQ / 64) * (GDIM / 16);          // MT=4
        gemm_tn_bf16<4><<<waveTiles / 8, 256, 0, stream>>>(
            Xe_bf, Weih, enc_b_ih, enc_b_hh, XpreE, SEQ, GDIM, HDIM, GDIM, GDIM);
        gemm_tn_bf16<4><<<waveTiles / 8, 256, 0, stream>>>(
            Xd_bf, Wdih, dec_b_ih, dec_b_hh, XpreD, SEQ, GDIM, HDIM, GDIM, GDIM);
    }

    // 4) encoder recurrence (persistent, 16 WGs, grid barrier per step)
    lstm_seq_kernel<<<16, 256, 0, stream>>>(XpreE, Wehh, h_glob, gbuf,
                                            enc_h0, enc_c0, (unsigned short*)nullptr,
                                            bar, bar + 1, SEQ);

    // 5) latent head: mean/logvar to d_out tail, dec initial h into h_glob
    latent_kernel<<<1, 256, 0, stream>>>(h_glob, mean_W, mean_b, logv_W, logv_b,
                                         lc_W, lc_b, c_in,
                                         out + MEAN_OFF, out + LOGV_OFF, h_glob);

    // 6) decoder recurrence, capturing all h_t as bf16
    lstm_seq_kernel<<<16, 256, 0, stream>>>(XpreD, Wdhh, h_glob, gbuf,
                                            (const float*)nullptr, (const float*)nullptr,
                                            Hdec, bar, bar + 1, SEQ);

    // 7) vocabulary projection: 26.4 GFLOP bf16 WMMA GEMM into d_out
    {
        int waveTiles = (SEQ / 64) * (NPAD / 16);          // MT=4 -> 12568
        gemm_tn_bf16<4><<<waveTiles / 8, 256, 0, stream>>>(
            Hdec, Wout, out_b, (const float*)nullptr, out + DIST_OFF,
            SEQ, NPAD, HDIM, VOCAB, VOCAB);
    }

    // 8) in-place log-softmax + argmax per step
    logsoftmax_argmax_kernel<<<SEQ, 256, 0, stream>>>(out + DIST_OFF, out);
}
